// LinearAttention_58686433132601
// MI455X (gfx1250) — matrix operand order repaired and checked by reading, not compiled, not run on hardware
//
#include <hip/hip_runtime.h>
#include <hip/hip_bf16.h>

typedef __attribute__((ext_vector_type(16))) _Float16 v16h;
typedef __attribute__((ext_vector_type(8)))  _Float16 v8h;
typedef __attribute__((ext_vector_type(4)))  _Float16 v4h;
typedef __attribute__((ext_vector_type(8)))  float    v8f;

union FragU { v16h v; v8h h[2]; };

// CDNA5 16-bit A fragment (16x32, M x K) from LDS tile laid out [row][k], row stride ldk halves.
// Lane L<16: row=L, K 0..7 in V0-3, K 16..23 in V4-7. Lane L>=16: row=L-16, K 8..15 / 24..31.
__device__ __forceinline__ v16h load_frag_a_lds(const _Float16* base, int ldk, int lane) {
  int row = lane & 15;
  int kb  = (lane < 16) ? 0 : 8;
  const _Float16* p = base + row * ldk + kb;
  FragU u;
  u.h[0] = *(const v8h*)(p);
  u.h[1] = *(const v8h*)(p + 16);
  return u.v;
}

// CDNA5 16-bit B fragment (32x16, K x N) from LDS tile laid out TRANSPOSED [col][k], stride ldk halves.
// Lane L<16: col=L, K 0..7 / 16..23; lane L>=16: col=L-16, K 8..15 / 24..31.
__device__ __forceinline__ v16h load_frag_b_lds(const _Float16* base, int ldk, int lane) {
  int col = lane & 15;
  int kb  = (lane < 16) ? 0 : 8;
  const _Float16* p = base + col * ldk + kb;
  FragU u;
  u.h[0] = *(const v8h*)(p);
  u.h[1] = *(const v8h*)(p + 16);
  return u.v;
}

__device__ __forceinline__ v8f wmma_f16(v16h a, v16h b, v8f c) {
  return __builtin_amdgcn_wmma_f32_16x16x32_f16(false, a, false, b, (short)0, c, false, false);
}

// ---------------------------------------------------------------------------
// Tiled WMMA GEMM: C[M x N] = A[M x K] * B[K x N]  (f16 compute, f32 accum)
// Block tile 128x128, 256 threads = 8 waves in a 4(M) x 2(N) grid,
// each wave owns 2x4 16x16 tiles. K stepped by 32.
// Double-buffered LDS + register-staged global prefetch: next tile's global
// loads are issued BEFORE the current tile's WMMAs so HBM latency overlaps
// the matrix math; one barrier per K-step. global_prefetch at distance 2.
// MODE 0: f16 output, elu(x)+1 applied to columns < eluLimit (qkv kernel)
// MODE 1: f32 output with bias add (projection kernel)
// ---------------------------------------------------------------------------
template<typename AT, int MODE>
__global__ __launch_bounds__(256)
void gemm_wmma_kernel(const void* __restrict__ Aptr, const float* __restrict__ Bm,
                      const float* __restrict__ bias, void* __restrict__ Cout,
                      int K, int lda, int ldb, int ldc, int eluLimit) {
  __shared__ __align__(16) _Float16 aT[2][128 * 40];
  __shared__ __align__(16) _Float16 bT[2][128 * 40];

  const int t = threadIdx.x;
  const int lane = t & 31;
  const int wid = t >> 5;
  const int waveM = wid >> 1;     // 0..3
  const int waveN = wid & 1;      // 0..1
  const int rowBase = blockIdx.x * 128;
  const int colBase = blockIdx.y * 128;

  // staging registers (one K-step of A and B per thread)
  float4 aRegF[4];     // f32 A path
  v8h    aRegH[2];     // f16 A path
  float4 bRegF[4];

  auto loadA = [&](int ks) {
    if (sizeof(AT) == 4) {
      const float* A = (const float*)Aptr;
      #pragma unroll
      for (int i = 0; i < 4; ++i) {
        int idx = t + i * 256;          // 0..1023 float4 chunks (128 rows x 8)
        int row = idx >> 3;
        int c4  = (idx & 7) << 2;
        aRegF[i] = *(const float4*)&A[(size_t)(rowBase + row) * lda + ks + c4];
      }
    } else {
      const _Float16* A = (const _Float16*)Aptr;
      #pragma unroll
      for (int i = 0; i < 2; ++i) {
        int idx = t + i * 256;          // 0..511 v8h chunks (128 rows x 4)
        int row = idx >> 2;
        int c8  = (idx & 3) << 3;
        aRegH[i] = *(const v8h*)&A[(size_t)(rowBase + row) * lda + ks + c8];
      }
    }
  };
  auto loadB = [&](int ks) {
    #pragma unroll
    for (int i = 0; i < 4; ++i) {
      int idx = t + i * 256;            // 0..1023 float4 chunks (32 k x 32)
      int kk = idx >> 5;                // 0..31
      int n4 = (idx & 31) << 2;         // 0..124
      bRegF[i] = *(const float4*)&Bm[(size_t)(ks + kk) * ldb + colBase + n4];
    }
  };
  auto storeA = [&](int buf) {
    if (sizeof(AT) == 4) {
      #pragma unroll
      for (int i = 0; i < 4; ++i) {
        int idx = t + i * 256;
        int row = idx >> 3;
        int c4  = (idx & 7) << 2;
        v4h hv;
        hv[0] = (_Float16)aRegF[i].x; hv[1] = (_Float16)aRegF[i].y;
        hv[2] = (_Float16)aRegF[i].z; hv[3] = (_Float16)aRegF[i].w;
        *(v4h*)&aT[buf][row * 40 + c4] = hv;
      }
    } else {
      #pragma unroll
      for (int i = 0; i < 2; ++i) {
        int idx = t + i * 256;
        int row = idx >> 2;
        int c8  = (idx & 3) << 3;
        *(v8h*)&aT[buf][row * 40 + c8] = aRegH[i];
      }
    }
  };
  auto storeB = [&](int buf) {
    #pragma unroll
    for (int i = 0; i < 4; ++i) {
      int idx = t + i * 256;
      int kk = idx >> 5;
      int n4 = (idx & 31) << 2;
      bT[buf][(n4 + 0) * 40 + kk] = (_Float16)bRegF[i].x;
      bT[buf][(n4 + 1) * 40 + kk] = (_Float16)bRegF[i].y;
      bT[buf][(n4 + 2) * 40 + kk] = (_Float16)bRegF[i].z;
      bT[buf][(n4 + 3) * 40 + kk] = (_Float16)bRegF[i].w;
    }
  };

  v8f acc[2][4];
  #pragma unroll
  for (int i = 0; i < 2; ++i)
    #pragma unroll
    for (int j = 0; j < 4; ++j)
      #pragma unroll
      for (int q = 0; q < 8; ++q) acc[i][j][q] = 0.0f;

  const int nSteps = K >> 5;

  // prologue: stage tile 0 into buffer 0
  loadA(0);
  loadB(0);
  storeA(0);
  storeB(0);
  __syncthreads();

  for (int s = 0; s < nSteps; ++s) {
    const int buf = s & 1;
    const int ks1 = (s + 1) << 5;

    // issue next tile's global loads first -> overlap HBM latency with WMMAs
    if (s + 1 < nSteps) {
      loadA(ks1);
      loadB(ks1);
    }
    // prefetch tile s+2 toward L2/WGP$
    if (s + 2 < nSteps) {
      const int ks2 = (s + 2) << 5;
      if (sizeof(AT) == 4) {
        __builtin_prefetch(&((const float*)Aptr)[(size_t)(rowBase + (t >> 3)) * lda + ks2 + ((t & 7) << 2)], 0, 1);
      } else {
        __builtin_prefetch(&((const _Float16*)Aptr)[(size_t)(rowBase + (t >> 2)) * lda + ks2 + ((t & 3) << 3)], 0, 1);
      }
      __builtin_prefetch(&Bm[(size_t)(ks2 + (t >> 5)) * ldb + colBase + ((t & 31) << 2)], 0, 1);
    }

    // compute current buffer
    v16h af[2], bf[4];
    #pragma unroll
    for (int i = 0; i < 2; ++i)
      af[i] = load_frag_a_lds(aT[buf] + (waveM * 32 + i * 16) * 40, 40, lane);
    #pragma unroll
    for (int j = 0; j < 4; ++j)
      bf[j] = load_frag_b_lds(bT[buf] + (waveN * 64 + j * 16) * 40, 40, lane);
    #pragma unroll
    for (int i = 0; i < 2; ++i)
      #pragma unroll
      for (int j = 0; j < 4; ++j)
        acc[i][j] = wmma_f16(af[i], bf[j], acc[i][j]);

    // convert + park next tile into the other LDS buffer
    if (s + 1 < nSteps) {
      storeA(buf ^ 1);
      storeB(buf ^ 1);
    }
    __syncthreads();
  }

  // ---- epilogue ----
  const int colL = lane & 15;
  const int rOff = (lane < 16) ? 0 : 8;
  #pragma unroll
  for (int i = 0; i < 2; ++i) {
    int growBase = rowBase + waveM * 32 + i * 16 + rOff;
    #pragma unroll
    for (int j = 0; j < 4; ++j) {
      int gcol = colBase + waveN * 64 + j * 16 + colL;
      #pragma unroll
      for (int v = 0; v < 8; ++v) {
        float g = acc[i][j][v];
        size_t grow = (size_t)(growBase + v);
        if (MODE == 0) {
          if (gcol < eluLimit) g = (g > 0.0f) ? (g + 1.0f) : __expf(g);  // elu+1
          ((_Float16*)Cout)[grow * ldc + gcol] = (_Float16)g;
        } else {
          ((float*)Cout)[grow * ldc + gcol] = g + bias[gcol];
        }
      }
    }
  }
}

// ---------------------------------------------------------------------------
// kv state: kv[head][d][e] = sum_n kf[n][d]*v[n][e]; ksum[head][d] = sum_n kf[n][d]
// Split-K over 8 chunks of 1024 n per head; f32 atomics finish the reduction.
// 8 waves each own 2 of the 16 (d,e) 16x16 tiles; kf/v staged transposed in LDS.
// ---------------------------------------------------------------------------
__global__ __launch_bounds__(256)
void kv_state_kernel(const _Float16* __restrict__ qkv, float* __restrict__ kvBuf,
                     float* __restrict__ ksumBuf) {
  __shared__ __align__(16) _Float16 ldsKf[64 * 40];   // [d][n]
  __shared__ __align__(16) _Float16 ldsV [64 * 40];   // [e][n]
  __shared__ float ldsKs[64];

  const int t = threadIdx.x, lane = t & 31, wid = t >> 5;
  const int head = blockIdx.x >> 3;
  const int sp   = blockIdx.x & 7;
  const int b = head >> 3, h = head & 7;
  const size_t rowBaseG = (size_t)b * 8192 + sp * 1024;

  if (t < 64) ldsKs[t] = 0.0f;

  const int tm = wid >> 1;          // d tile 0..3
  const int e0 = (wid & 1) * 32;    // e columns base (2 tiles)
  v8f acc[2];
  #pragma unroll
  for (int j = 0; j < 2; ++j)
    #pragma unroll
    for (int q = 0; q < 8; ++q) acc[j][q] = 0.0f;

  float ksacc[8] = {0, 0, 0, 0, 0, 0, 0, 0};
  const int nl = t >> 3;            // n within 32-tile (fixed per thread)
  const int d8 = (t & 7) << 3;      // 8 d's per thread (fixed)

  for (int it = 0; it < 32; ++it) {
    __syncthreads();
    size_t row = rowBaseG + it * 32 + nl;
    const _Float16* pk = qkv + row * 1536 + 512  + h * 64 + d8;
    const _Float16* pv = qkv + row * 1536 + 1024 + h * 64 + d8;
    v8h k8 = *(const v8h*)pk;
    v8h v8 = *(const v8h*)pv;
    #pragma unroll
    for (int j = 0; j < 8; ++j) {
      ldsKf[(d8 + j) * 40 + nl] = k8[j];
      ldsV [(d8 + j) * 40 + nl] = v8[j];
      ksacc[j] += (float)k8[j];
    }
    __syncthreads();
    v16h af  = load_frag_a_lds(ldsKf + tm * 16 * 40, 40, lane);
    v16h bf0 = load_frag_b_lds(ldsV  + (e0 + 0)  * 40, 40, lane);
    v16h bf1 = load_frag_b_lds(ldsV  + (e0 + 16) * 40, 40, lane);
    acc[0] = wmma_f16(af, bf0, acc[0]);
    acc[1] = wmma_f16(af, bf1, acc[1]);
  }

  // ksum: LDS reduce then one global atomic per d
  #pragma unroll
  for (int j = 0; j < 8; ++j) atomicAdd(&ldsKs[d8 + j], ksacc[j]);
  __syncthreads();
  if (t < 64) atomicAdd(&ksumBuf[head * 64 + t], ldsKs[t]);

  // kv split-K reduction via global f32 atomics
  const int colL = lane & 15, rOff = (lane < 16) ? 0 : 8;
  float* kvp = kvBuf + (size_t)head * 4096;
  #pragma unroll
  for (int j = 0; j < 2; ++j) {
    int e  = e0 + j * 16 + colL;
    int d0 = tm * 16 + rOff;
    #pragma unroll
    for (int v = 0; v < 8; ++v) atomicAdd(&kvp[(d0 + v) * 64 + e], acc[j][v]);
  }
}

// ---------------------------------------------------------------------------
// attn: out[n][e] = (qf[n] . kv[:,e]) / (qf[n] . ksum + eps), per head.
// Block = 128 rows x 64 cols; kv staged as f16 B^T in LDS; A frags straight
// from global (contiguous 16B chunks per lane). Normalizer z in LDS.
// ---------------------------------------------------------------------------
__global__ __launch_bounds__(256)
void attn_kernel(const _Float16* __restrict__ qkv, const float* __restrict__ kvBuf,
                 const float* __restrict__ ksumBuf, _Float16* __restrict__ attnOut) {
  __shared__ __align__(16) _Float16 ldsKV[64 * 72];   // [e][d]
  __shared__ float ldsKs[64];
  __shared__ float ldsZ[128];

  const int t = threadIdx.x, lane = t & 31, wid = t >> 5;
  const int head = blockIdx.y;
  const int b = head >> 3, h = head & 7;
  const int rBase = blockIdx.x * 128;

  { // stage kv transposed, f32 -> f16
    int e  = t >> 2;
    int dB = (t & 3) * 16;
    const float* kvp = kvBuf + (size_t)head * 4096;
    #pragma unroll
    for (int j = 0; j < 16; ++j) {
      int d = dB + j;
      ldsKV[e * 72 + d] = (_Float16)kvp[d * 64 + e];
    }
  }
  if (t < 64) ldsKs[t] = ksumBuf[head * 64 + t];
  __syncthreads();

  if (t < 128) { // normalizer z = qf . ksum
    size_t row = (size_t)b * 8192 + rBase + t;
    const _Float16* pq = qkv + row * 1536 + h * 64;
    float z = 0.0f;
    #pragma unroll
    for (int c = 0; c < 8; ++c) {
      v8h q8 = *(const v8h*)(pq + c * 8);
      #pragma unroll
      for (int j = 0; j < 8; ++j) z += (float)q8[j] * ldsKs[c * 8 + j];
    }
    ldsZ[t] = z + 1e-6f;
  }
  __syncthreads();

  v8f acc[4];
  #pragma unroll
  for (int j = 0; j < 4; ++j)
    #pragma unroll
    for (int q = 0; q < 8; ++q) acc[j][q] = 0.0f;

  #pragma unroll
  for (int s = 0; s < 2; ++s) {   // K = 64 = 2 x 32
    int rowL = wid * 16 + (lane & 15);
    size_t row = (size_t)b * 8192 + rBase + rowL;
    int kb = s * 32 + ((lane < 16) ? 0 : 8);
    const _Float16* p = qkv + row * 1536 + h * 64 + kb;
    FragU u;
    u.h[0] = *(const v8h*)p;
    u.h[1] = *(const v8h*)(p + 16);
    v16h af = u.v;
    #pragma unroll
    for (int j = 0; j < 4; ++j) {
      v16h bf = load_frag_b_lds(ldsKV + j * 16 * 72 + s * 32, 72, lane);
      acc[j] = wmma_f16(af, bf, acc[j]);
    }
  }

  const int colL = lane & 15, rOff = (lane < 16) ? 0 : 8;
  #pragma unroll
  for (int j = 0; j < 4; ++j) {
    int e = j * 16 + colL;
    #pragma unroll
    for (int v = 0; v < 8; ++v) {
      int mL = wid * 16 + rOff + v;
      float val = acc[j][v] / ldsZ[mL];
      attnOut[((size_t)b * 8192 + rBase + mL) * 512 + h * 64 + e] = (_Float16)val;
    }
  }
}

__global__ void zero_kernel(float* p, int n) {
  int i = blockIdx.x * 256 + threadIdx.x;
  if (i < n) p[i] = 0.0f;
}

extern "C" void kernel_launch(void* const* d_in, const int* in_sizes, int n_in,
                              void* d_out, int out_size, void* d_ws, size_t ws_size,
                              hipStream_t stream) {
  const float* x      = (const float*)d_in[0];   // [8,8192,512]
  const float* w_qkv  = (const float*)d_in[1];   // [512,1536]
  const float* w_proj = (const float*)d_in[2];   // [512,512]
  const float* b_proj = (const float*)d_in[3];   // [512]
  float* out = (float*)d_out;                    // [65536,512]

  char* ws = (char*)d_ws;
  const size_t QKV_BYTES  = (size_t)65536 * 1536 * 2;   // 201326592
  const size_t ATTN_BYTES = (size_t)65536 * 512 * 2;    //  67108864
  const size_t KV_BYTES   = (size_t)64 * 4096 * 4;      //   1048576
  _Float16* qkvF16  = (_Float16*)ws;
  _Float16* attnF16 = (_Float16*)(ws + QKV_BYTES);
  float*    kvBuf   = (float*)(ws + QKV_BYTES + ATTN_BYTES);
  float*    ksumBuf = (float*)(ws + QKV_BYTES + ATTN_BYTES + KV_BYTES);

  // 1) qkv = x @ w_qkv, elu+1 fused on q/k columns (<1024), f16 out
  gemm_wmma_kernel<float, 0><<<dim3(512, 12), 256, 0, stream>>>(
      x, w_qkv, nullptr, qkvF16, /*K=*/512, /*lda=*/512, /*ldb=*/1536, /*ldc=*/1536,
      /*eluLimit=*/1024);

  // 2) zero kv + ksum accumulators (contiguous: 64*4096 + 64*64 floats)
  zero_kernel<<<(266240 + 255) / 256, 256, 0, stream>>>(kvBuf, 266240);

  // 3) kv state + ksum (split-K 8 per head, atomic reduction)
  kv_state_kernel<<<dim3(64 * 8), 256, 0, stream>>>(qkvF16, kvBuf, ksumBuf);

  // 4) attn = (qf @ kv) / (qf . ksum + eps), f16 out
  attn_kernel<<<dim3(64, 64), 256, 0, stream>>>(qkvF16, kvBuf, ksumBuf, attnF16);

  // 5) y = attn @ w_proj + b_proj, f32 out
  gemm_wmma_kernel<_Float16, 1><<<dim3(512, 4), 256, 0, stream>>>(
      attnF16, w_proj, b_proj, out, /*K=*/512, /*lda=*/512, /*ldb=*/512, /*ldc=*/512,
      /*eluLimit=*/0);
}
